// TemporalPropagator_82343112999146
// MI455X (gfx1250) — matrix operand order repaired and checked by reading, not compiled, not run on hardware
//
#include <hip/hip_runtime.h>

#define DIM   1024
#define BATCH 8192

typedef __bf16 bf16_t;
typedef __attribute__((ext_vector_type(16))) __bf16 v16bf;
typedef __attribute__((ext_vector_type(8)))  __bf16 v8bf;
typedef __attribute__((ext_vector_type(8)))  float  v8f;
typedef int v4i_t __attribute__((vector_size(4 * sizeof(int))));

#define WMMA_BF16(a, b, c) \
  __builtin_amdgcn_wmma_f32_16x16x32_bf16(false, (a), false, (b), (short)0, (c), false, false)

#ifndef __has_builtin
#define __has_builtin(x) 0
#endif

// ---- CDNA5 async memory->LDS copy (GLOBAL_LOAD_ASYNC_TO_LDS_B128, ASYNCcnt) ----
__device__ __forceinline__ void async_cp16(const void* g, void* l) {
#if __has_builtin(__builtin_amdgcn_global_load_async_to_lds_b128)
  __builtin_amdgcn_global_load_async_to_lds_b128(
      (__attribute__((address_space(1))) v4i_t*)(unsigned long long)g,
      (__attribute__((address_space(3))) v4i_t*)(unsigned int)(unsigned long long)l,
      0, 0);
#else
  // VDST = LDS byte address (low 32 bits of generic pointer), VADDR = 64-bit global address
  asm volatile("global_load_async_to_lds_b128 %0, %1, off"
               :: "v"((unsigned int)(unsigned long long)l), "v"(g)
               : "memory");
#endif
}
__device__ __forceinline__ void wait_async0() {
#if __has_builtin(__builtin_amdgcn_s_wait_asynccnt)
  __builtin_amdgcn_s_wait_asynccnt(0);
#else
  asm volatile("s_wait_asynccnt 0x0" ::: "memory");
#endif
}

// ---- fragment loaders (wave32, 16-bit WMMA VGPR layouts from CDNA5 ISA 7.12.2) ----
// A 16x32: lane L: m = L%16, khalf = L/16; data = K[8*khalf .. +7] and K[8*khalf+16 .. +7]
__device__ __forceinline__ v16bf frag_a(const bf16_t* base, int ld, int lane) {
  const int idx = lane & 15, half = lane >> 4;
  const bf16_t* p = base + idx * ld + half * 8;
  union { v16bf v; v8bf h[2]; } u;
  u.h[0] = *(const v8bf*)(p);
  u.h[1] = *(const v8bf*)(p + 16);
  return u.v;
}
// B 32x16 (Bt stored N x K row-major): lane L: n = L%16, data = K[8*half .. +7] and K[8*half+16 .. +7]
__device__ __forceinline__ v16bf frag_b(const bf16_t* base, int ld, int lane) {
  const int idx = lane & 15, half = lane >> 4;
  const bf16_t* p = base + idx * ld + half * 8;
  union { v16bf v; v8bf h[2]; } u;
  u.h[0] = *(const v8bf*)(p);
  u.h[1] = *(const v8bf*)(p + 16);
  return u.v;
}

// ---- generic (complex) WMMA GEMM:  C[m,n] = sum_k A[m,k] * (Br[n,k] + i*sgn*Bi[n,k]) ----
// Block tile 128x64x32, 8 waves (4x2), wave tile 32x32. Double-buffered LDS fed by
// async-to-LDS copies; ASYNCcnt wait + workgroup barrier per K-step.
template <bool ACPLX, bool BCPLX>
__global__ __launch_bounds__(256) void gemm_wmma_bf16(
    const bf16_t* __restrict__ Ar, const bf16_t* __restrict__ Ai,
    const bf16_t* __restrict__ Br, const bf16_t* __restrict__ Bi,   // Bt layout: N x K row-major
    float* __restrict__ Cre, float* __restrict__ Cim,               // f32 plane outputs (nullable)
    bf16_t* __restrict__ Cre16, bf16_t* __restrict__ Cim16,         // bf16 plane outputs (nullable)
    const float* __restrict__ addRe, const float* __restrict__ addIm, // accumulate source (nullable)
    const float* __restrict__ bias,                                 // per-column real bias (nullable)
    float* __restrict__ outInter,                                   // interleaved complex out (nullable)
    float sgn, int M, int N, int K, int lda, int ldb, int ldc, int ldc16)
{
  constexpr int BM = 128, BN = 64, BK = 32, BKP = BK + 8;
  __shared__ __attribute__((aligned(16))) bf16_t sAr[2][BM * BKP];
  __shared__ __attribute__((aligned(16))) bf16_t sAi[ACPLX ? 2 : 1][ACPLX ? BM * BKP : 16];
  __shared__ __attribute__((aligned(16))) bf16_t sBr[2][BN * BKP];
  __shared__ __attribute__((aligned(16))) bf16_t sBi[BCPLX ? 2 : 1][BCPLX ? BN * BKP : 16];

  const int tid  = threadIdx.x;
  const int lane = tid & 31;
  const int wave = tid >> 5;
  const int wm   = wave & 3;   // 0..3 (M direction)
  const int wn   = wave >> 2;  // 0..1 (N direction)
  const int bm0  = blockIdx.y * BM;
  const int bn0  = blockIdx.x * BN;

  // stage one K-tile into LDS buffer `buf` with async copies (16B per lane)
  auto stage = [&](int k0, int buf) {
#pragma unroll
    for (int i = 0; i < 2; ++i) {          // A: 512 chunks of 8 bf16, 2 per thread
      const int c8 = i * 256 + tid;
      const int r = c8 >> 2, c = (c8 & 3) * 8;
      const long go = (long)(bm0 + r) * lda + k0 + c;
      async_cp16(&Ar[go], &sAr[buf][r * BKP + c]);
      if constexpr (ACPLX) async_cp16(&Ai[go], &sAi[buf][r * BKP + c]);
    }
    {                                      // B: 256 chunks, 1 per thread
      const int r = tid >> 2, c = (tid & 3) * 8;
      const long go = (long)(bn0 + r) * ldb + k0 + c;
      async_cp16(&Br[go], &sBr[buf][r * BKP + c]);
      if constexpr (BCPLX) async_cp16(&Bi[go], &sBi[buf][r * BKP + c]);
    }
  };

  v8f acc_rr[2][2] = {};
  v8f acc_ri[2][2] = {};
  v8f acc_ir[2][2] = {};
  v8f acc_ii[2][2] = {};

  const int ksteps = K / BK;
  stage(0, 0);
  wait_async0();
  __syncthreads();

  for (int kt = 0; kt < ksteps; ++kt) {
    const int cur = kt & 1;
    if (kt + 1 < ksteps) stage((kt + 1) * BK, cur ^ 1);   // prefetch next tile (async)

    v16bf ar[2], ai[2], br[2], bi[2];
#pragma unroll
    for (int mi = 0; mi < 2; ++mi) {
      ar[mi] = frag_a(&sAr[cur][(wm * 32 + mi * 16) * BKP], BKP, lane);
      if constexpr (ACPLX) ai[mi] = frag_a(&sAi[cur][(wm * 32 + mi * 16) * BKP], BKP, lane);
    }
#pragma unroll
    for (int ni = 0; ni < 2; ++ni) {
      br[ni] = frag_b(&sBr[cur][(wn * 32 + ni * 16) * BKP], BKP, lane);
      if constexpr (BCPLX) bi[ni] = frag_b(&sBi[cur][(wn * 32 + ni * 16) * BKP], BKP, lane);
    }
#pragma unroll
    for (int mi = 0; mi < 2; ++mi)
#pragma unroll
      for (int ni = 0; ni < 2; ++ni) {
        acc_rr[mi][ni] = WMMA_BF16(ar[mi], br[ni], acc_rr[mi][ni]);
        if constexpr (BCPLX) acc_ri[mi][ni] = WMMA_BF16(ar[mi], bi[ni], acc_ri[mi][ni]);
        if constexpr (ACPLX && BCPLX) {
          acc_ir[mi][ni] = WMMA_BF16(ai[mi], br[ni], acc_ir[mi][ni]);
          acc_ii[mi][ni] = WMMA_BF16(ai[mi], bi[ni], acc_ii[mi][ni]);
        }
      }

    wait_async0();       // my async writes into buf^1 have landed in LDS
    __syncthreads();     // everyone done reading buf / writing buf^1
  }

  // epilogue: C layout — lane n = L%16, VGPR r -> m = r + 8*(L/16)
  const int idx = lane & 15, half = lane >> 4;
#pragma unroll
  for (int mi = 0; mi < 2; ++mi)
#pragma unroll
    for (int ni = 0; ni < 2; ++ni) {
      const int gm0 = bm0 + wm * 32 + mi * 16 + half * 8;
      const int gn  = bn0 + wn * 32 + ni * 16 + idx;
#pragma unroll
      for (int r = 0; r < 8; ++r) {
        const int m = gm0 + r;
        float cre, cim;
        if constexpr (ACPLX && BCPLX) {
          cre = acc_rr[mi][ni][r] - sgn * acc_ii[mi][ni][r];
          cim = acc_ir[mi][ni][r] + sgn * acc_ri[mi][ni][r];
        } else if constexpr (BCPLX) {
          cre = acc_rr[mi][ni][r];
          cim = acc_ri[mi][ni][r];
        } else {
          cre = acc_rr[mi][ni][r];
          cim = 0.f;
        }
        const long co = (long)m * ldc + gn;
        if (addRe) cre += addRe[co];
        if (BCPLX && addIm) cim += addIm[co];
        if (bias) cre += bias[gn];
        if (Cre) Cre[co] = cre;
        if (Cim) Cim[co] = cim;
        const long co16 = (long)m * ldc16 + gn;
        if (Cre16) Cre16[co16] = (bf16_t)cre;
        if (Cim16) Cim16[co16] = (bf16_t)cim;
        if (outInter) {
          const long o = ((long)m * N + gn) * 2;
          outInter[o]     = cre;
          outInter[o + 1] = cim;
        }
      }
    }
}

// ---- A = 0.5(R - R^T) + i*0.5(I + I^T); emit P=A, Pt=A^T (bf16) and Tt=I+A^T (f32+bf16) ----
__global__ void k_build(const float* __restrict__ re, const float* __restrict__ im,
                        bf16_t* Pr, bf16_t* Pi, bf16_t* Ptr_, bf16_t* Pti,
                        float* Ttre, float* Ttim, bf16_t* Ttr16, bf16_t* Tti16) {
  const int i = blockIdx.x * 256 + threadIdx.x;
  if (i >= DIM * DIM) return;
  const int m = i >> 10, n = i & 1023;
  const float are = 0.5f * (re[i] - re[n * DIM + m]);   // antisymmetric
  const float aim = 0.5f * (im[i] + im[n * DIM + m]);   // symmetric
  Pr[i] = (bf16_t)are;      Pi[i] = (bf16_t)aim;
  Ptr_[i] = (bf16_t)(-are); Pti[i] = (bf16_t)aim;       // A^T[m,n] = -are + i*aim
  const float ttre = (m == n ? 1.f : 0.f) - are;        // (I+A)^T
  Ttre[i] = ttre;  Ttim[i] = aim;
  Ttr16[i] = (bf16_t)ttre; Tti16[i] = (bf16_t)aim;
}

// ---- U = 2*Tt^T - I ; also U scaled per-column by exp(ssign*log_sigma) ----
__global__ void k_finish(const float* __restrict__ Ttre, const float* __restrict__ Ttim,
                         const float* __restrict__ log_sigma, float ssign,
                         bf16_t* Ur, bf16_t* Ui, bf16_t* Usr, bf16_t* Usi) {
  const int i = blockIdx.x * 256 + threadIdx.x;
  if (i >= DIM * DIM) return;
  const int m = i >> 10, n = i & 1023;
  const float ure = 2.f * Ttre[n * DIM + m] - (m == n ? 1.f : 0.f);
  const float uim = 2.f * Ttim[n * DIM + m];
  Ur[i] = (bf16_t)ure; Ui[i] = (bf16_t)uim;
  const float sc = __expf(ssign * log_sigma[n]);
  Usr[i] = (bf16_t)(ure * sc); Usi[i] = (bf16_t)(uim * sc);
}

__global__ void k_cvt(const float* __restrict__ s, bf16_t* __restrict__ d, long n) {
  const long i = (long)blockIdx.x * 256 + threadIdx.x;
  if (i < n) d[i] = (bf16_t)s[i];
}

// ---- exponential-integrator pointwise update ----
__global__ void k_pointwise(const float* __restrict__ htre, const float* __restrict__ htim,
                            const float* __restrict__ xtre, const float* __restrict__ xtim,
                            const float* __restrict__ delta, const float* __restrict__ dt,
                            const float* __restrict__ ld, const float* __restrict__ lf,
                            const float* __restrict__ dts, const float* __restrict__ src_re,
                            const float* __restrict__ src_im, const float* __restrict__ law_re,
                            const float* __restrict__ law_im,
                            bf16_t* __restrict__ hnr, bf16_t* __restrict__ hni) {
  const long i = (long)blockIdx.x * 256 + threadIdx.x;
  if (i >= (long)BATCH * DIM) return;
  const int n = (int)(i & (DIM - 1));
  const int b = (int)(i >> 10);
  const float dre  = delta[(long)b * 2048 + n];
  const float dim_ = delta[(long)b * 2048 + DIM + n];
  const float lre = -__expf(ld[n]) + law_re[n] + dre;
  const float lim = lf[n] + law_im[n] + dim_;
  const float dtb = dt[b];
  const float dtn = dtb / dts[n];            // DT_REF_BASE = 1
  const float zr = lre * dtn, zi = lim * dtn;
  const float r2 = zr * zr + zi * zi;
  const float er = __expf(zr), cy = __cosf(zi), sy = __sinf(zi);
  const float eZr = er * cy, eZi = er * sy;
  float p1r, p1i;
  if (r2 < 1e-8f) {                          // |Z| < 1e-4
    p1r = 1.f + 0.5f * zr + (zr * zr - zi * zi) * (1.f / 6.f);
    p1i = 0.5f * zi + (zr * zi) * (1.f / 3.f);
  } else {
    const float nr = eZr - 1.f, ni = eZi, inv = 1.f / r2;
    p1r = (nr * zr + ni * zi) * inv;         // (e^Z - 1)/Z
    p1i = (ni * zr - nr * zi) * inv;
  }
  const float fr = p1r * dtb, fi = p1i * dtb;  // phi1 * (dt_n * dt_ref_eff) = phi1 * dt
  const float hr = htre[i], hi = htim[i];
  const float xr = xtre[i] + src_re[n], xi = xtim[i] + src_im[n];
  hnr[i] = (bf16_t)(hr * eZr - hi * eZi + xr * fr - xi * fi);
  hni[i] = (bf16_t)(hr * eZi + hi * eZr + xr * fi + xi * fr);
}

extern "C" void kernel_launch(void* const* d_in, const int* in_sizes, int n_in,
                              void* d_out, int out_size, void* d_ws, size_t ws_size,
                              hipStream_t stream) {
  (void)in_sizes; (void)n_in; (void)out_size;
  const float* h_prev    = (const float*)d_in[0];
  const float* x_input   = (const float*)d_in[1];
  const float* dt        = (const float*)d_in[2];
  const float* u_raw_re  = (const float*)d_in[3];
  const float* u_raw_im  = (const float*)d_in[4];
  const float* v_raw_re  = (const float*)d_in[5];
  const float* v_raw_im  = (const float*)d_in[6];
  const float* log_sigma = (const float*)d_in[7];
  const float* ld_       = (const float*)d_in[8];
  const float* lf        = (const float*)d_in[9];
  const float* dt_scaler = (const float*)d_in[10];
  const float* src_re    = (const float*)d_in[11];
  const float* src_im    = (const float*)d_in[12];
  const float* law_re    = (const float*)d_in[13];
  const float* law_im    = (const float*)d_in[14];
  const float* sel_w     = (const float*)d_in[15];
  const float* sel_b     = (const float*)d_in[16];
  float* out = (float*)d_out;

  const size_t d2 = (size_t)DIM * DIM;
  const size_t Bd = (size_t)BATCH * DIM;
  char* base = (char*)d_ws;
  size_t off = 0;
  auto alloc = [&](size_t bytes) -> void* {
    void* p = base + off;
    off = (off + bytes + 255) & ~(size_t)255;
    return p;
  };
  // bf16 matrix buffers
  bf16_t *P[2][2], *Pt[2][2], *Tt16[2][2];
  for (int b = 0; b < 2; ++b)
    for (int c = 0; c < 2; ++c) {
      P[b][c]    = (bf16_t*)alloc(d2 * 2);
      Pt[b][c]   = (bf16_t*)alloc(d2 * 2);
      Tt16[b][c] = (bf16_t*)alloc(d2 * 2);
    }
  bf16_t* U16r = (bf16_t*)alloc(d2 * 2); bf16_t* U16i = (bf16_t*)alloc(d2 * 2);
  bf16_t* Usr  = (bf16_t*)alloc(d2 * 2); bf16_t* Usi  = (bf16_t*)alloc(d2 * 2);
  bf16_t* V16r = (bf16_t*)alloc(d2 * 2); bf16_t* V16i = (bf16_t*)alloc(d2 * 2);
  bf16_t* Vsr  = (bf16_t*)alloc(d2 * 2); bf16_t* Vsi_ = (bf16_t*)alloc(d2 * 2);
  bf16_t* Minvr = (bf16_t*)alloc(d2 * 2); bf16_t* Minvi = (bf16_t*)alloc(d2 * 2);
  bf16_t* M16r  = (bf16_t*)alloc(d2 * 2); bf16_t* M16i  = (bf16_t*)alloc(d2 * 2);
  bf16_t* h16    = (bf16_t*)alloc(Bd * 2);
  bf16_t* x16    = (bf16_t*)alloc(Bd * 2);
  bf16_t* xcat   = (bf16_t*)alloc(Bd * 2 * 2);          // BATCH x 2048
  bf16_t* selw16 = (bf16_t*)alloc(d2 * 4 * 2);          // 2048 x 2048
  bf16_t* hnr16  = (bf16_t*)alloc(Bd * 2);
  bf16_t* hni16  = (bf16_t*)alloc(Bd * 2);
  // f32 buffers
  float* Ttre = (float*)alloc(d2 * 4);
  float* Ttim = (float*)alloc(d2 * 4);
  float* htre = (float*)alloc(Bd * 4);
  float* htim = (float*)alloc(Bd * 4);
  float* xtre = (float*)alloc(Bd * 4);
  float* xtim = (float*)alloc(Bd * 4);
  float* delta = (float*)alloc(Bd * 2 * 4);             // BATCH x 2048
  if (off > ws_size) return;  // workspace too small

  const dim3 blk(256);
  const dim3 gMM(DIM / 64, DIM / 128);       // 1024^3 GEMMs
  const dim3 gB(DIM / 64, BATCH / 128);      // 8192x1024 GEMMs
  const dim3 gSel(2 * DIM / 64, BATCH / 128);
  const int eG_d2 = (int)(d2 / 256), eG_Bd = (int)(Bd / 256);

  const float* raws[2][2] = {{u_raw_re, u_raw_im}, {v_raw_re, v_raw_im}};
  bf16_t* Uo[2][2]  = {{U16r, U16i}, {V16r, V16i}};
  bf16_t* Uso[2][2] = {{Usr, Usi}, {Vsr, Vsi_}};
  const float ssgn[2] = {1.f, -1.f};  // U scaled by exp(+ls), V by exp(-ls)

  for (int mt = 0; mt < 2; ++mt) {
    k_build<<<eG_d2, blk, 0, stream>>>(raws[mt][0], raws[mt][1],
        P[0][0], P[0][1], Pt[0][0], Pt[0][1], Ttre, Ttim, Tt16[0][0], Tt16[0][1]);
    int pc = 0, tc = 0;
    for (int lvl = 0; lvl < 4; ++lvl) {       // builds sum_{j<32} A^j via Pi(I + A^{2^k})
      const int np = pc ^ 1, tn = tc ^ 1;
      gemm_wmma_bf16<true, true><<<gMM, blk, 0, stream>>>(
          P[pc][0], P[pc][1], Pt[pc][0], Pt[pc][1],
          nullptr, nullptr, P[np][0], P[np][1],
          nullptr, nullptr, nullptr, nullptr,
          1.f, DIM, DIM, DIM, DIM, DIM, DIM, DIM);
      gemm_wmma_bf16<true, true><<<gMM, blk, 0, stream>>>(
          Pt[pc][0], Pt[pc][1], P[pc][0], P[pc][1],
          nullptr, nullptr, Pt[np][0], Pt[np][1],
          nullptr, nullptr, nullptr, nullptr,
          1.f, DIM, DIM, DIM, DIM, DIM, DIM, DIM);
      gemm_wmma_bf16<true, true><<<gMM, blk, 0, stream>>>(   // Tt <- (T + P*T)^T
          Tt16[tc][0], Tt16[tc][1], P[np][0], P[np][1],
          Ttre, Ttim, Tt16[tn][0], Tt16[tn][1],
          Ttre, Ttim, nullptr, nullptr,
          1.f, DIM, DIM, DIM, DIM, DIM, DIM, DIM);
      pc = np; tc = tn;
    }
    k_finish<<<eG_d2, blk, 0, stream>>>(Ttre, Ttim, log_sigma, ssgn[mt],
        Uo[mt][0], Uo[mt][1], Uso[mt][0], Uso[mt][1]);
  }

  // M_inv = (V*s_inv) @ U^H ; M = (U*s) @ V^H   (conj via epilogue sign)
  gemm_wmma_bf16<true, true><<<gMM, blk, 0, stream>>>(
      Vsr, Vsi_, U16r, U16i, nullptr, nullptr, Minvr, Minvi,
      nullptr, nullptr, nullptr, nullptr, -1.f, DIM, DIM, DIM, DIM, DIM, DIM, DIM);
  gemm_wmma_bf16<true, true><<<gMM, blk, 0, stream>>>(
      Usr, Usi, V16r, V16i, nullptr, nullptr, M16r, M16i,
      nullptr, nullptr, nullptr, nullptr, -1.f, DIM, DIM, DIM, DIM, DIM, DIM, DIM);

  k_cvt<<<eG_Bd, blk, 0, stream>>>(h_prev, h16, (long)Bd);
  k_cvt<<<eG_Bd, blk, 0, stream>>>(x_input, x16, (long)Bd);
  k_cvt<<<(int)(d2 * 4 / 256), blk, 0, stream>>>(sel_w, selw16, (long)(d2 * 4));

  // encodes: X @ M_inv^T  (real A, complex B)
  gemm_wmma_bf16<false, true><<<gB, blk, 0, stream>>>(
      h16, nullptr, Minvr, Minvi, htre, htim, nullptr, nullptr,
      nullptr, nullptr, nullptr, nullptr, 1.f, BATCH, DIM, DIM, DIM, DIM, DIM, DIM);
  gemm_wmma_bf16<false, true><<<gB, blk, 0, stream>>>(    // also writes [re|im] into xcat
      x16, nullptr, Minvr, Minvi, xtre, xtim, xcat, xcat + DIM,
      nullptr, nullptr, nullptr, nullptr, 1.f, BATCH, DIM, DIM, DIM, DIM, DIM, 2 * DIM);

  // selector: delta = xcat @ sel_w^T + sel_b  (pure real)
  gemm_wmma_bf16<false, false><<<gSel, blk, 0, stream>>>(
      xcat, nullptr, selw16, nullptr, delta, nullptr, nullptr, nullptr,
      nullptr, nullptr, sel_b, nullptr, 1.f, BATCH, 2 * DIM, 2 * DIM,
      2 * DIM, 2 * DIM, 2 * DIM, 2 * DIM);

  k_pointwise<<<eG_Bd, blk, 0, stream>>>(htre, htim, xtre, xtim, delta, dt, ld_, lf,
      dt_scaler, src_re, src_im, law_re, law_im, hnr16, hni16);

  // out = h_next @ M^T  (complex x complex, interleaved complex store)
  gemm_wmma_bf16<true, true><<<gB, blk, 0, stream>>>(
      hnr16, hni16, M16r, M16i, nullptr, nullptr, nullptr, nullptr,
      nullptr, nullptr, nullptr, out, 1.f, BATCH, DIM, DIM, DIM, DIM, DIM, DIM);
}
